// GraphEmbedding_747324310157
// MI455X (gfx1250) — matrix operand order repaired and checked by reading, not compiled, not run on hardware
//
#include <hip/hip_runtime.h>

#define NN   4096          // nodes
#define TT   12            // timesteps
#define BB   64            // batch
#define CC   16            // channels
#define BC   (BB*CC)       // 1024
#define JJ   (BC*TT)       // 12288 flattened GEMM columns
#define NT   (NN*TT)       // 49152 per-channel plane
#define CNT  ((size_t)CC*NT)
#define KC   112           // 7*C concat channels
#define APT  10
#define XELEMS ((size_t)BB*CNT)   // 50,331,648

typedef __attribute__((ext_vector_type(16))) __bf16 v16bf;
typedef __attribute__((ext_vector_type(8)))  __bf16 v8bf;
typedef __attribute__((ext_vector_type(8)))  float  v8f;

static __device__ __forceinline__ v16bf cat8(v8bf lo, v8bf hi) {
  return __builtin_shufflevector(lo, hi,
         0,1,2,3,4,5,6,7,8,9,10,11,12,13,14,15);
}

// ---------------------------------------------------------------------------
// f32 -> bf16 bulk convert (8 elements / thread, b128 in / b128 out)
// ---------------------------------------------------------------------------
__launch_bounds__(256)
__global__ void cvt_bf16(const float* __restrict__ src,
                         __bf16* __restrict__ dst) {
  size_t p = ((size_t)blockIdx.x * 256 + threadIdx.x) * 8;
  if (p >= XELEMS) return;
  v8bf o;
  #pragma unroll
  for (int i = 0; i < 8; ++i) o[i] = (__bf16)src[p + i];
  *(v8bf*)(dst + p) = o;
}

// ---------------------------------------------------------------------------
// WMMA GEMM (bf16 in / bf16 out): Y[bc,m,t] = sum_n AT[m,n] * X[bc,n,t],
// columns j = bc*T + t. Workgroup = 8 waves, tile M=256 x J=32.
// Per wave: 2 m-tiles x 2 j-tiles (4 accumulators); K-tile 64, double-buffered
// LDS B-tile; per 32-deep sub-step: 2 A-frags + 2 B-frags feed 4 WMMAs.
// ---------------------------------------------------------------------------
__launch_bounds__(256)
__global__ void nconv_wmma(const __bf16* __restrict__ AT,
                           const __bf16* __restrict__ X,
                           __bf16* __restrict__ Y) {
  const int jt   = blockIdx.x;          // 0..JJ/32-1
  const int mb   = blockIdx.y;          // 0..NN/256-1
  const int tid  = threadIdx.x;
  const int wave = tid >> 5;
  const int lane = tid & 31;
  const int L    = lane & 15;
  const int h    = lane >> 4;

  const int j0 = jt * 32;
  const int m0 = mb * 256 + wave * 32;  // wave owns rows [m0, m0+32)

  // [buf][j][k] bf16, K-row padded to 72 halfwords (144B, 16B-aligned)
  __shared__ __bf16 ldsB[2][32][72];

  // --- staging map: thread (k8, jl) copies X(k0+k8+8u, j0+jl), u=0..7
  const int jl = tid & 31;
  const int k8 = tid >> 5;              // 0..7
  const int j  = j0 + jl;
  const int bc = j / TT;
  const int t  = j - bc * TT;
  const __bf16* xbase = X + (size_t)bc * NT + (size_t)k8 * TT + t;

  // --- A fragment pointers: rows m0+L and m0+16+L, chunk offset 16h
  const __bf16* aptr0 = AT + (size_t)(m0 + L) * NN + 16 * h;
  const __bf16* aptr1 = aptr0 + (size_t)16 * NN;

  // --- output maps for the two j-tiles
  const int jo0  = j0 + L;
  const int jo1  = j0 + 16 + L;
  const int bc0  = jo0 / TT, t0 = jo0 - bc0 * TT;
  const int bc1  = jo1 / TT, t1 = jo1 - bc1 * TT;
  __bf16* yp0 = Y + (size_t)bc0 * NT + t0;
  __bf16* yp1 = Y + (size_t)bc1 * NT + t1;

  v8f acc00 = {}, acc01 = {}, acc10 = {}, acc11 = {};

  // prologue: stage k-tile 0 -> buf 0
  #pragma unroll
  for (int u = 0; u < 8; ++u)
    ldsB[0][jl][k8 + 8 * u] = xbase[(size_t)(8 * u) * TT];

  const int KSTEPS = NN / 64;           // 64
  int buf = 0;
  for (int kk = 0; kk < KSTEPS; ++kk) {
    __syncthreads();                    // tile kk ready in ldsB[buf]
    if (kk + 1 < KSTEPS) {              // prefetch next tile into other buffer
      const __bf16* xp = xbase + (size_t)(kk + 1) * 64 * TT;
      #pragma unroll
      for (int u = 0; u < 8; ++u)
        ldsB[buf ^ 1][jl][k8 + 8 * u] = xp[(size_t)(8 * u) * TT];
    }
    #pragma unroll
    for (int s = 0; s < 2; ++s) {       // two 32-deep sub-steps
      // B fragments: col N=L (+16), K = 32s + 16h + [0..15], contiguous LDS
      const __bf16* bq0 = &ldsB[buf][L][32 * s + 16 * h];
      const __bf16* bq1 = &ldsB[buf][16 + L][32 * s + 16 * h];
      v16bf b0 = cat8(*(const v8bf*)bq0, *(const v8bf*)(bq0 + 8));
      v16bf b1 = cat8(*(const v8bf*)bq1, *(const v8bf*)(bq1 + 8));
      // A fragments: two contiguous 16B global loads each
      const __bf16* ap0 = aptr0 + kk * 64 + 32 * s;
      const __bf16* ap1 = aptr1 + kk * 64 + 32 * s;
      v16bf a0 = cat8(*(const v8bf*)ap0, *(const v8bf*)(ap0 + 8));
      acc00 = __builtin_amdgcn_wmma_f32_16x16x32_bf16(
                  false, a0, false, b0, (short)0, acc00, false, false);
      acc01 = __builtin_amdgcn_wmma_f32_16x16x32_bf16(
                  false, a0, false, b1, (short)0, acc01, false, false);
      v16bf a1 = cat8(*(const v8bf*)ap1, *(const v8bf*)(ap1 + 8));
      acc10 = __builtin_amdgcn_wmma_f32_16x16x32_bf16(
                  false, a1, false, b0, (short)0, acc10, false, false);
      acc11 = __builtin_amdgcn_wmma_f32_16x16x32_bf16(
                  false, a1, false, b1, (short)0, acc11, false, false);
    }
    buf ^= 1;
  }

  #pragma unroll
  for (int r = 0; r < 8; ++r) {
    int m = m0 + r + 8 * h;
    yp0[(size_t)m * TT]        = (__bf16)acc00[r];
    yp1[(size_t)m * TT]        = (__bf16)acc01[r];
    yp0[(size_t)(m + 16) * TT] = (__bf16)acc10[r];
    yp1[(size_t)(m + 16) * TT] = (__bf16)acc11[r];
  }
}

// ---------------------------------------------------------------------------
// 32x32 LDS transpose + f32->bf16: AT[m][n] = bf16(A[n][m])
// ---------------------------------------------------------------------------
__launch_bounds__(256)
__global__ void transpose_bf16(const float* __restrict__ A,
                               __bf16* __restrict__ AT) {
  __shared__ float tile[32][33];
  const int tx = threadIdx.x & 31;
  const int ty = threadIdx.x >> 5;      // 0..7
  const int n0 = blockIdx.y * 32;
  const int m0 = blockIdx.x * 32;
  #pragma unroll
  for (int i = 0; i < 4; ++i) {
    int r = ty + i * 8;
    tile[r][tx] = A[(size_t)(n0 + r) * NN + m0 + tx];
  }
  __syncthreads();
  #pragma unroll
  for (int i = 0; i < 4; ++i) {
    int r = ty + i * 8;
    AT[(size_t)(m0 + r) * NN + n0 + tx] = (__bf16)tile[tx][r];
  }
}

// ---------------------------------------------------------------------------
// Adaptive adjacency: AT[m][n] = bf16( softmax_m( relu(E1[n,:] @ E2[:,m]) ) )
// ---------------------------------------------------------------------------
__launch_bounds__(256)
__global__ void adp_softmax_T(const float* __restrict__ E1,
                              const float* __restrict__ E2,
                              __bf16* __restrict__ AT) {
  const int n   = blockIdx.x;
  const int tid = threadIdx.x;
  __shared__ float row[NN];
  __shared__ float red[256];

  float e1[APT];
  #pragma unroll
  for (int k = 0; k < APT; ++k) e1[k] = E1[n * APT + k];

  float lmax = -1e30f;
  for (int m = tid; m < NN; m += 256) {
    float d = 0.f;
    #pragma unroll
    for (int k = 0; k < APT; ++k) d += e1[k] * E2[(size_t)k * NN + m];
    d = fmaxf(d, 0.f);
    row[m] = d;
    lmax = fmaxf(lmax, d);
  }
  red[tid] = lmax;
  __syncthreads();
  for (int s = 128; s > 0; s >>= 1) {
    if (tid < s) red[tid] = fmaxf(red[tid], red[tid + s]);
    __syncthreads();
  }
  const float rmax = red[0];
  __syncthreads();

  float lsum = 0.f;
  for (int m = tid; m < NN; m += 256) {
    float e = __expf(row[m] - rmax);
    row[m] = e;
    lsum += e;
  }
  red[tid] = lsum;
  __syncthreads();
  for (int s = 128; s > 0; s >>= 1) {
    if (tid < s) red[tid] += red[tid + s];
    __syncthreads();
  }
  const float inv = 1.0f / red[0];
  for (int m = tid; m < NN; m += 256)
    AT[(size_t)m * NN + n] = (__bf16)(row[m] * inv);
}

// ---------------------------------------------------------------------------
// out = x + bias + W[:,0:16] @ x   (identity branch of the concat + residual)
// ---------------------------------------------------------------------------
__launch_bounds__(256)
__global__ void init_out(const float* __restrict__ x,
                         const float* __restrict__ W,
                         const float* __restrict__ bias,
                         float* __restrict__ out) {
  __shared__ float Wl[CC][CC];
  __shared__ float bl[CC];
  {
    int c = threadIdx.x >> 4, k = threadIdx.x & 15;
    Wl[c][k] = W[c * KC + k];
    if (threadIdx.x < CC) bl[threadIdx.x] = bias[threadIdx.x];
  }
  __syncthreads();
  size_t p = (size_t)blockIdx.x * 256 + threadIdx.x;
  if (p >= (size_t)BB * NT) return;
  size_t bidx = p / NT, r = p % NT;
  const float* xp = x + bidx * CNT + r;
  float* op = out + bidx * CNT + r;
  float xv[CC];
  #pragma unroll
  for (int c = 0; c < CC; ++c) xv[c] = xp[(size_t)c * NT];
  #pragma unroll
  for (int c = 0; c < CC; ++c) {
    float acc = xv[c] + bl[c];
    #pragma unroll
    for (int k = 0; k < CC; ++k) acc += Wl[c][k] * xv[k];
    op[(size_t)c * NT] = acc;
  }
}

// ---------------------------------------------------------------------------
// out += W[:,koff:koff+16] @ y1  +  W[:,koff+16:koff+32] @ y2   (y in bf16)
// ---------------------------------------------------------------------------
__launch_bounds__(256)
__global__ void accum_out(const __bf16* __restrict__ y1,
                          const __bf16* __restrict__ y2,
                          const float* __restrict__ W, int koff,
                          float* __restrict__ out) {
  __shared__ float W1[CC][CC], W2[CC][CC];
  {
    int c = threadIdx.x >> 4, k = threadIdx.x & 15;
    W1[c][k] = W[c * KC + koff + k];
    W2[c][k] = W[c * KC + koff + CC + k];
  }
  __syncthreads();
  size_t p = (size_t)blockIdx.x * 256 + threadIdx.x;
  if (p >= (size_t)BB * NT) return;
  size_t bidx = p / NT, r = p % NT;
  const __bf16* y1p = y1 + bidx * CNT + r;
  const __bf16* y2p = y2 + bidx * CNT + r;
  float* op = out + bidx * CNT + r;
  float v1[CC], v2[CC];
  #pragma unroll
  for (int c = 0; c < CC; ++c) {
    v1[c] = (float)y1p[(size_t)c * NT];
    v2[c] = (float)y2p[(size_t)c * NT];
  }
  #pragma unroll
  for (int c = 0; c < CC; ++c) {
    float acc = op[(size_t)c * NT];
    #pragma unroll
    for (int k = 0; k < CC; ++k) acc += W1[c][k] * v1[k] + W2[c][k] * v2[k];
    op[(size_t)c * NT] = acc;
  }
}

// ---------------------------------------------------------------------------
extern "C" void kernel_launch(void* const* d_in, const int* in_sizes, int n_in,
                              void* d_out, int out_size, void* d_ws, size_t ws_size,
                              hipStream_t stream) {
  const float* x   = (const float*)d_in[0];
  const float* s0  = (const float*)d_in[1];
  const float* s1  = (const float*)d_in[2];
  const float* e1  = (const float*)d_in[3];
  const float* e2  = (const float*)d_in[4];
  const float* W   = (const float*)d_in[5];
  const float* bia = (const float*)d_in[6];
  float* out = (float*)d_out;

  char* ws = (char*)d_ws;
  __bf16* AT  = (__bf16*)ws;  ws += (size_t)NN * NN * sizeof(__bf16);  // 33.5 MB
  __bf16* xbf = (__bf16*)ws;  ws += XELEMS * sizeof(__bf16);           // 100.7 MB
  __bf16* y1  = (__bf16*)ws;  ws += XELEMS * sizeof(__bf16);           // 100.7 MB
  __bf16* y2  = (__bf16*)ws;                                           // 100.7 MB

  const dim3 gGemm(JJ / 32, NN / 256);        // 384 x 16
  const dim3 gTr(NN / 32, NN / 32);           // 128 x 128
  const unsigned gElem = (unsigned)(((size_t)BB * NT + 255) / 256);
  const unsigned gCvt  = (unsigned)((XELEMS / 8 + 255) / 256);

  // bf16 copy of x (L2-resident GEMM operand) + identity/bias/residual branch
  cvt_bf16<<<gCvt, 256, 0, stream>>>(x, xbf);
  init_out<<<gElem, 256, 0, stream>>>(x, W, bia, out);

  // support0: two diffusion hops
  transpose_bf16<<<gTr, 256, 0, stream>>>(s0, AT);
  nconv_wmma<<<gGemm, 256, 0, stream>>>(AT, xbf, y1);
  nconv_wmma<<<gGemm, 256, 0, stream>>>(AT, y1,  y2);
  accum_out<<<gElem, 256, 0, stream>>>(y1, y2, W, 16, out);

  // support1
  transpose_bf16<<<gTr, 256, 0, stream>>>(s1, AT);
  nconv_wmma<<<gGemm, 256, 0, stream>>>(AT, xbf, y1);
  nconv_wmma<<<gGemm, 256, 0, stream>>>(AT, y1,  y2);
  accum_out<<<gElem, 256, 0, stream>>>(y1, y2, W, 48, out);

  // adaptive support: softmax(relu(E1@E2)) written transposed as bf16
  adp_softmax_T<<<NN, 256, 0, stream>>>(e1, e2, AT);
  nconv_wmma<<<gGemm, 256, 0, stream>>>(AT, xbf, y1);
  nconv_wmma<<<gGemm, 256, 0, stream>>>(AT, y1,  y2);
  accum_out<<<gElem, 256, 0, stream>>>(y1, y2, W, 80, out);
}
